// DeepNNDendroMatrix_9680856285494
// MI455X (gfx1250) — matrix operand order repaired and checked by reading, not compiled, not run on hardware
//
#include <hip/hip_runtime.h>

// Problem constants (from reference)
#define Bsz 2048
#define Fsz 512
#define Hsz 256
#define Esz 128
#define NNODES 4096

typedef __attribute__((ext_vector_type(16))) __bf16 v16bf;
typedef __attribute__((ext_vector_type(8)))  __bf16 v8bf;
typedef __attribute__((ext_vector_type(8)))  float  v8f;

static __device__ __forceinline__ v8f wmma_bf16(v16bf a, v16bf b, v8f c) {
  return __builtin_amdgcn_wmma_f32_16x16x32_bf16(false, a, false, b, (short)0, c,
                                                 false, false);
}

// ---------------------------------------------------------------------------
// 1) colsT[b][e] = path_mat[e][node_idx[b]]
__global__ void k_gather_cols(const float* __restrict__ path_mat,
                              const int* __restrict__ node_idx,
                              float* __restrict__ colsT) {
  int b = blockIdx.x;
  int e = threadIdx.x;                 // 128 threads
  int n = node_idx[b];
  colsT[b * Esz + e] = path_mat[(size_t)e * NNODES + n];
}

// 2) xb = bf16(x)
__global__ void k_cvt_x(const float* __restrict__ x, __bf16* __restrict__ xb) {
  int i = blockIdx.x * 256 + threadIdx.x;
  xb[i] = (__bf16)x[i];
}

// 3) root0t[o][i] = bf16(root0[i][o])   (N-major layout for WMMA B operand)
__global__ void k_cvt_root0(const float* __restrict__ root0,
                            __bf16* __restrict__ r0t) {
  int tid = blockIdx.x * 256 + threadIdx.x;  // F*H threads
  int i = tid & (Fsz - 1);
  int o = tid >> 9;
  r0t[(size_t)o * Fsz + i] = (__bf16)root0[(size_t)i * Hsz + o];
}

// 4) Wb2[(o*E+e)][i] = bf16(delta0[o][i][e]) ; LDS-tiled 32x32 transpose
__global__ void k_tr_delta0(const float* __restrict__ delta0,
                            __bf16* __restrict__ Wb2) {
  __shared__ float tile[32][33];
  int o  = blockIdx.z;
  int i0 = blockIdx.x * 32;
  int e0 = blockIdx.y * 32;
  int tx = threadIdx.x & 31;
  int ty = threadIdx.x >> 5;  // 0..7
#pragma unroll
  for (int r = 0; r < 4; ++r) {
    int il = ty + r * 8;
    tile[il][tx] = delta0[((size_t)o * Fsz + i0 + il) * Esz + e0 + tx];
  }
  __syncthreads();
#pragma unroll
  for (int r = 0; r < 4; ++r) {
    int el = ty + r * 8;
    Wb2[((size_t)o * Esz + e0 + el) * Fsz + i0 + tx] = (__bf16)tile[tx][el];
  }
}

// Build A-operand in contiguous k order (B loads use the same):
// lane<16: K = {0..15}; lane>=16: K = {16..31}
static __device__ __forceinline__ v16bf load_A(const __bf16* __restrict__ rowp,
                                               int k0, int akb) {
  v8bf lo = *(const v8bf*)(rowp + k0 + akb);
  v8bf hi = *(const v8bf*)(rowp + k0 + 8 + akb);
  return __builtin_shufflevector(lo, hi, 0, 1, 2, 3, 4, 5, 6, 7, 8, 9, 10, 11,
                                 12, 13, 14, 15);
}

// ---------------------------------------------------------------------------
// 5) base = xb(2048x512) @ root0t^T -> (2048x256) f32, WMMA bf16
//    grid (B/64, H/32), 256 threads (8 waves), 16x16 tile per wave
__global__ void k_base_gemm(const __bf16* __restrict__ xb,
                            const __bf16* __restrict__ r0t,
                            float* __restrict__ base) {
  int w = threadIdx.x >> 5, lane = threadIdx.x & 31;
  int wm = w >> 1, wn = w & 1;
  int m0 = blockIdx.x * 64 + wm * 16;
  int n0 = blockIdx.y * 32 + wn * 16;
  int hi = lane & 16;
  int akb = hi ? 16 : 0, bkb = hi ? 16 : 0;
  const __bf16* aR = xb + (size_t)(m0 + (lane & 15)) * Fsz;
  const __bf16* bR = r0t + (size_t)(n0 + (lane & 15)) * Fsz;
  v8f c{};
  for (int kk = 0; kk < Fsz / 32; ++kk) {
    int k0 = kk * 32;
    v16bf a = load_A(aR, k0, akb);
    v16bf b = *(const v16bf*)(bR + k0 + bkb);
    c = wmma_bf16(a, b, c);
  }
  int col = n0 + (lane & 15);
#pragma unroll
  for (int v = 0; v < 8; ++v) {
    int mr = m0 + v + (hi ? 8 : 0);
    base[(size_t)mr * Hsz + col] = c[v];
  }
}

// ---------------------------------------------------------------------------
// 6) Main fused GEMM: G = xb @ Wb2^T over the 128 columns (e) of one o,
//    epilogue multiplies by colsT and reduces over e -> h1 = relu(base + .)
//    grid (B/128, H), 256 threads = 8 waves laid out 4(m) x 2(n);
//    each wave owns a 32(m) x 64(n) register tile = 8 WMMA accumulators,
//    so each B tile load feeds two WMMAs (better FLOP/byte than 16x64).
__global__ void k_main_gemm(const __bf16* __restrict__ xb,
                            const __bf16* __restrict__ Wb2,
                            const float* __restrict__ colsT,
                            const float* __restrict__ base,
                            float* __restrict__ h1) {
  __shared__ float scols[128 * Esz];   // 64 KB: cols for this block's 128 rows
  __shared__ float partial[128];
  int t = threadIdx.x;
  int b0 = blockIdx.x * 128;
  int o = blockIdx.y;

  for (int i = t; i < 128 * Esz; i += 256)
    scols[i] = colsT[(size_t)b0 * Esz + i];
  if (t < 128) partial[t] = 0.f;
  __syncthreads();

  int w = t >> 5, lane = t & 31;
  int wm = w >> 1, wn = w & 1;       // 4 m-slices (32 rows) x 2 n-halves (64)
  int hi = lane & 16;
  int akb = hi ? 16 : 0, bkb = hi ? 16 : 0;
  const __bf16* aR0 = xb + (size_t)(b0 + wm * 32 + (lane & 15)) * Fsz;
  const __bf16* aR1 = aR0 + (size_t)16 * Fsz;
  size_t nbase = (size_t)o * Esz + wn * 64 + (lane & 15);

  v8f c[2][4] = {{v8f{}, v8f{}, v8f{}, v8f{}}, {v8f{}, v8f{}, v8f{}, v8f{}}};
  for (int kk = 0; kk < Fsz / 32; ++kk) {
    int k0 = kk * 32;
    v16bf a0 = load_A(aR0, k0, akb);
    v16bf a1 = load_A(aR1, k0, akb);
#pragma unroll
    for (int tt = 0; tt < 4; ++tt) {
      v16bf b = *(const v16bf*)(Wb2 + (nbase + tt * 16) * Fsz + k0 + bkb);
      c[0][tt] = wmma_bf16(a0, b, c[0][tt]);
      c[1][tt] = wmma_bf16(a1, b, c[1][tt]);
    }
  }

  // Epilogue: weight by cols, reduce over e (16 lanes per half hold 16 cols)
  float s[2][8] = {{0.f, 0.f, 0.f, 0.f, 0.f, 0.f, 0.f, 0.f},
                   {0.f, 0.f, 0.f, 0.f, 0.f, 0.f, 0.f, 0.f}};
#pragma unroll
  for (int tt = 0; tt < 4; ++tt) {
    int e = wn * 64 + tt * 16 + (lane & 15);
#pragma unroll
    for (int mi = 0; mi < 2; ++mi) {
      int mlb = wm * 32 + mi * 16 + (hi ? 8 : 0);
#pragma unroll
      for (int v = 0; v < 8; ++v)
        s[mi][v] += c[mi][tt][v] * scols[(mlb + v) * Esz + e];
    }
  }
#pragma unroll
  for (int mask = 8; mask >= 1; mask >>= 1) {
#pragma unroll
    for (int mi = 0; mi < 2; ++mi)
#pragma unroll
      for (int v = 0; v < 8; ++v) s[mi][v] += __shfl_xor(s[mi][v], mask, 32);
  }
  if ((lane & 15) == 0) {
#pragma unroll
    for (int mi = 0; mi < 2; ++mi) {
      int mlb = wm * 32 + mi * 16 + (hi ? 8 : 0);
#pragma unroll
      for (int v = 0; v < 8; ++v) atomicAdd(&partial[mlb + v], s[mi][v]);
    }
  }
  __syncthreads();
  if (t < 128) {
    int b = b0 + t;
    h1[(size_t)b * Hsz + o] =
        fmaxf(0.f, base[(size_t)b * Hsz + o] + partial[t]);
  }
}

// ---------------------------------------------------------------------------
// 7) Layer 2 (O==1): out[b] = relu( sum_i h1[b,i]*(root1[i] + delta1[i,:].cols) )
__global__ void k_layer2(const float* __restrict__ h1,
                         const float* __restrict__ colsT,
                         const float* __restrict__ root1,
                         const float* __restrict__ delta1,
                         float* __restrict__ out) {
  __shared__ float sc[Esz];
  __shared__ float red[256];
  int b = blockIdx.x;
  int t = threadIdx.x;  // = i, 256 threads
  if (t < Esz) sc[t] = colsT[(size_t)b * Esz + t];
  __syncthreads();
  float wv = root1[t];
  const float* dp = delta1 + (size_t)t * Esz;
#pragma unroll 8
  for (int e = 0; e < Esz; ++e) wv += dp[e] * sc[e];
  red[t] = h1[(size_t)b * Hsz + t] * wv;
  __syncthreads();
  for (int s = 128; s > 0; s >>= 1) {
    if (t < s) red[t] += red[t + s];
    __syncthreads();
  }
  if (t == 0) out[b] = fmaxf(0.f, red[0]);
}

// ---------------------------------------------------------------------------
extern "C" void kernel_launch(void* const* d_in, const int* in_sizes, int n_in,
                              void* d_out, int out_size, void* d_ws,
                              size_t ws_size, hipStream_t stream) {
  (void)in_sizes; (void)n_in; (void)out_size; (void)ws_size;
  const float* x        = (const float*)d_in[0];
  const int*   node_idx = (const int*)d_in[1];
  const float* path_mat = (const float*)d_in[2];
  const float* root0    = (const float*)d_in[3];
  const float* root1    = (const float*)d_in[4];
  const float* delta0   = (const float*)d_in[5];
  const float* delta1   = (const float*)d_in[6];
  float* out = (float*)d_out;

  // Workspace carve (all 1KB-aligned): total ~41.2 MB
  char* ws = (char*)d_ws;
  float*  colsT = (float*)(ws);                               // B*E*4   = 1 MiB
  __bf16* xb    = (__bf16*)(ws + (1u << 20));                 // B*F*2   = 2 MiB
  __bf16* r0t   = (__bf16*)(ws + (3u << 20));                 // F*H*2   = 256 KiB
  __bf16* Wb2   = (__bf16*)(ws + (3u << 20) + (1u << 18));    // F*H*E*2 = 32 MiB
  float*  base  = (float*)(ws + (35u << 20) + (1u << 18));    // B*H*4   = 2 MiB
  float*  h1    = (float*)(ws + (37u << 20) + (1u << 18));    // B*H*4   = 2 MiB

  k_gather_cols<<<Bsz, Esz, 0, stream>>>(path_mat, node_idx, colsT);
  k_cvt_x<<<(Bsz * Fsz) / 256, 256, 0, stream>>>(x, xb);
  k_cvt_root0<<<(Fsz * Hsz) / 256, 256, 0, stream>>>(root0, r0t);
  k_tr_delta0<<<dim3(Fsz / 32, Esz / 32, Hsz), 256, 0, stream>>>(delta0, Wb2);
  k_base_gemm<<<dim3(Bsz / 64, Hsz / 32), 256, 0, stream>>>(xb, r0t, base);
  k_main_gemm<<<dim3(Bsz / 128, Hsz), 256, 0, stream>>>(xb, Wb2, colsT, base,
                                                        h1);
  k_layer2<<<Bsz, 256, 0, stream>>>(h1, colsT, root1, delta1, out);
}
